// MultiHeadAttention_59021440582086
// MI455X (gfx1250) — matrix operand order repaired and checked by reading, not compiled, not run on hardware
//
#include <hip/hip_runtime.h>
#include <hip/hip_bf16.h>

// ---------------------------------------------------------------------------
// MI455X (gfx1250) implementation, round 3.
// ~160 GFLOP, 97% in six dense GEMMs -> bf16 WMMA, fp32 accumulate.
// Round-3 change: weights are pre-transposed + pre-converted to bf16 once
// (N-major Wt[n][k], ~4 MB total) so the GEMM B-tile staging is a pure
// contiguous bf16 copy (2x b128 load + 2x b128 LDS store per thread) instead
// of fp32 loads + 16 cvt + 16 scalar ds_store_b16 transposed scatter per
// thread per k-step. WMMA inner loop (8 wmma / k-step per wave) unchanged.
// ---------------------------------------------------------------------------

#define Bc   8
#define Nc   512
#define Dc   512
#define Hc   16
#define DHc  32
#define HHc  8
#define DEGc 16
#define Ec   (Bc * Nc * DEGc)

#define LN2F 0.6931471805599453f

typedef __bf16 bf16_t;
typedef bf16_t v16bf __attribute__((ext_vector_type(16)));
typedef bf16_t v8bf  __attribute__((ext_vector_type(8)));
typedef float  v8f   __attribute__((ext_vector_type(8)));

union FragU { v16bf v; v8bf h[2]; };

__device__ __forceinline__ float softplus_m_ln2(float x) {
  float r;
  if (x > 20.0f)       r = x;
  else if (x < -20.0f) r = __expf(x);
  else                 r = __logf(1.0f + __expf(x));
  return r - LN2F;
}

// ---------------------------------------------------------------------------
// One-time weight prep: W (K x N, f32, row-major) -> Wt (N x K, bf16).
// Coalesced reads along n; ~2.1M elements total across all weights.
// ---------------------------------------------------------------------------
__global__ void transpose_w_kernel(const float* __restrict__ W,
                                   bf16_t* __restrict__ Wt, int K, int N)
{
  int idx = blockIdx.x * 256 + threadIdx.x;
  if (idx < K * N) {
    int k = idx / N, n = idx % N;
    Wt[(long)n * K + k] = (bf16_t)W[idx];
  }
}

// ---------------------------------------------------------------------------
// Tiled bf16-WMMA GEMM: C = act(A @ B + bias), B given transposed bf16.
// Block 256 threads (8 waves), block tile 128x128, K-step 32, double-buffered.
// Wave grid 4(M) x 2(N); each wave owns 32x64 = 2x4 WMMA 16x16 tiles.
// Requires M % 128 == 0, N % 128 == 0, K % 32 == 0 (true for this workload).
// GATHER: logical A row m = A[idx1[m]][k] (k<splitK) else A[idx2[m]][k-splitK].
// ---------------------------------------------------------------------------
#define BMT 128
#define BNT 128
#define BKT 32
#define PADT 40   // LDS row stride in bf16 (80B: 16B aligned, conflict-free)

template<bool ABF16, bool GATHER, bool ACT, bool CBF16>
__global__ __launch_bounds__(256)
void gemm_wmma(const void* __restrict__ Aptr, long lda,
               const bf16_t* __restrict__ Bt, long ldbt,   // N x K bf16
               const float* __restrict__ bias,
               void* __restrict__ Cptr, long ldc,
               int K,
               const int* __restrict__ idx1, const int* __restrict__ idx2,
               int splitK)
{
  __shared__ __attribute__((aligned(16))) bf16_t sA[2][BMT][PADT];
  __shared__ __attribute__((aligned(16))) bf16_t sB[2][BNT][PADT];

  const int t    = threadIdx.x;
  const int lane = t & 31;
  const int w    = t >> 5;     // 0..7
  const int wm   = w >> 1;     // 0..3
  const int wn   = w & 1;      // 0..1
  const int m0   = blockIdx.y * BMT;
  const int n0   = blockIdx.x * BNT;

  // loader mappings: each thread moves 16 A elems + 16 B elems per k-step
  const int ar = t >> 1;           // A row 0..127
  const int ac = (t & 1) * 16;     // A col base {0,16}
  const int br = t >> 1;           // B n-row 0..127 (of transposed Wt)
  const int bc = (t & 1) * 16;     // B k base {0,16}

  // gather rows resolved once, outside the k-loop
  const int gm = m0 + ar;
  int rowI = gm, rowJ = gm;
  if (GATHER) { rowI = idx1[gm]; rowJ = idx2[gm]; }

  v8f acc[2][4] = {};

  // staging registers (global -> regs -> LDS)
  float fA[16];
  v8bf  hA[2];
  v8bf  hB[2];

  auto loadA = [&](int k0) {
    int kc  = k0 + ac;
    int row = rowI;
    if (GATHER) { if (kc >= splitK) { row = rowJ; kc -= splitK; } }
    if (ABF16) {
      const v8bf* p = (const v8bf*)((const bf16_t*)Aptr + (long)row * lda + kc);
      hA[0] = p[0];
      hA[1] = p[1];
    } else {
      const float4* p = (const float4*)((const float*)Aptr + (long)row * lda + kc);
      #pragma unroll
      for (int q = 0; q < 4; ++q) {
        float4 v = p[q];
        fA[q * 4 + 0] = v.x; fA[q * 4 + 1] = v.y;
        fA[q * 4 + 2] = v.z; fA[q * 4 + 3] = v.w;
      }
    }
  };

  auto loadB = [&](int k0) {
    const v8bf* p = (const v8bf*)(Bt + (long)(n0 + br) * ldbt + k0 + bc);
    hB[0] = p[0];
    hB[1] = p[1];
  };

  auto stage = [&](int db) {
    v8bf o0, o1;
    if (ABF16) { o0 = hA[0]; o1 = hA[1]; }
    else {
      #pragma unroll
      for (int i = 0; i < 8; ++i) { o0[i] = (bf16_t)fA[i]; o1[i] = (bf16_t)fA[8 + i]; }
    }
    *(v8bf*)&sA[db][ar][ac]     = o0;
    *(v8bf*)&sA[db][ar][ac + 8] = o1;
    *(v8bf*)&sB[db][br][bc]     = hB[0];
    *(v8bf*)&sB[db][br][bc + 8] = hB[1];
  };

  auto compute = [&](int db) {
    FragU af[2], bf_[4];
    const int arow = wm * 32 + (lane & 15);
    const int kg   = (lane >> 4) * 16;
    #pragma unroll
    for (int si = 0; si < 2; ++si) {
      const bf16_t* p = &sA[db][arow + si * 16][kg];
      af[si].h[0] = *(const v8bf*)(p);
      af[si].h[1] = *(const v8bf*)(p + 8);
    }
    const int ncol = wn * 64 + (lane & 15);
    const int kc2  = (lane >> 4) * 16;
    #pragma unroll
    for (int sj = 0; sj < 4; ++sj) {
      const bf16_t* p = &sB[db][ncol + sj * 16][kc2];
      bf_[sj].h[0] = *(const v8bf*)(p);
      bf_[sj].h[1] = *(const v8bf*)(p + 8);
    }
    #pragma unroll
    for (int si = 0; si < 2; ++si)
      #pragma unroll
      for (int sj = 0; sj < 4; ++sj)
        acc[si][sj] = __builtin_amdgcn_wmma_f32_16x16x32_bf16(
            false, af[si].v, false, bf_[sj].v,
            (short)0, acc[si][sj], false, false);
  };

  // ---- pipelined main loop ----
  loadA(0); loadB(0);
  stage(0);
  __syncthreads();
  const int nk = K / BKT;
  for (int kt = 0; kt < nk; ++kt) {
    const int db = kt & 1;
    if (kt + 1 < nk) { loadA((kt + 1) * BKT); loadB((kt + 1) * BKT); }
    compute(db);
    if (kt + 1 < nk) stage(db ^ 1);
    __syncthreads();
  }

  // ---- epilogue: lane l, reg r -> m = (l>>4)*8 + r, n = l&15 ----
  const int mbase = m0 + wm * 32;
  const int nbase = n0 + wn * 64;
  #pragma unroll
  for (int si = 0; si < 2; ++si) {
    #pragma unroll
    for (int sj = 0; sj < 4; ++sj) {
      const int n  = nbase + sj * 16 + (lane & 15);
      const int mr = mbase + si * 16 + ((lane >> 4) * 8);
      const float bv = bias[n];
      #pragma unroll
      for (int r = 0; r < 8; ++r) {
        float v = acc[si][sj][r] + bv;
        if (ACT) v = softplus_m_ln2(v);
        if (CBF16) ((bf16_t*)Cptr)[(long)(mr + r) * ldc + n] = (bf16_t)v;
        else       ((float*) Cptr)[(long)(mr + r) * ldc + n] = v;
      }
    }
  }
}

// ---------------------------------------------------------------------------
// Edge gather index build: row = b*N + i / b*N + j
// ---------------------------------------------------------------------------
__global__ void build_idx_kernel(const int* __restrict__ pb,
                                 const int* __restrict__ pi,
                                 const int* __restrict__ pj,
                                 int* __restrict__ ri, int* __restrict__ rj)
{
  int e = blockIdx.x * 256 + threadIdx.x;
  if (e < Ec) {
    int base = pb[e] * Nc;
    ri[e] = base + pi[e];
    rj[e] = base + pj[e];
  }
}

// ---------------------------------------------------------------------------
// Fused global attention (heads 0..7). One block per (b, h, q-row).
// ---------------------------------------------------------------------------
__global__ __launch_bounds__(256)
void global_attn_kernel(const float* __restrict__ qb, const float* __restrict__ kb,
                        const float* __restrict__ vb, const unsigned char* __restrict__ mask,
                        float* __restrict__ ctx, float* __restrict__ topsc)
{
  const int bid = blockIdx.x;
  const int r = bid & (Nc - 1);
  const int h = (bid >> 9) & (HHc - 1);
  const int b = bid >> 12;
  const int t = threadIdx.x;
  const float scale = 0.17677669529663687f; // 1/sqrt(32)

  __shared__ float qrow[DHc];
  __shared__ float sc[Nc];
  __shared__ float red[256];
  __shared__ float part[8][DHc];

  if (t < DHc) qrow[t] = qb[((long)(b * Nc + r)) * Dc + h * DHc + t] * scale;
  __syncthreads();

  for (int j = t; j < Nc; j += 256) {
    const float* kr = kb + ((long)(b * Nc + j)) * Dc + h * DHc;
    float s = 0.0f;
    #pragma unroll
    for (int d = 0; d < DHc; ++d) s += qrow[d] * kr[d];
    if (h == 0) topsc[((long)(b * Nc + r)) * Nc + j] = s; // pre-mask copy
    if (mask[((long)(b * Nc + r)) * Nc + j]) s = -1e18f;
    sc[j] = s;
  }
  __syncthreads();

  red[t] = fmaxf(sc[t], sc[t + 256]);
  __syncthreads();
  for (int s2 = 128; s2 > 0; s2 >>= 1) {
    if (t < s2) red[t] = fmaxf(red[t], red[t + s2]);
    __syncthreads();
  }
  const float mx = red[0];
  __syncthreads();

  float e0 = __expf(sc[t] - mx), e1 = __expf(sc[t + 256] - mx);
  sc[t] = e0; sc[t + 256] = e1;
  red[t] = e0 + e1;
  __syncthreads();
  for (int s2 = 128; s2 > 0; s2 >>= 1) {
    if (t < s2) red[t] += red[t + s2];
    __syncthreads();
  }
  const float inv = 1.0f / red[0];
  __syncthreads();

  {
    const int d = t & (DHc - 1);
    const int g = t >> 5;
    float a = 0.0f;
    for (int j = g * 64; j < g * 64 + 64; ++j)
      a += sc[j] * vb[((long)(b * Nc + j)) * Dc + h * DHc + d];
    part[g][d] = a;
  }
  __syncthreads();
  if (t < DHc) {
    float s = 0.0f;
    #pragma unroll
    for (int g = 0; g < 8; ++g) s += part[g][t];
    ctx[((long)(b * Nc + r)) * Dc + h * DHc + t] = s * inv;
  }
}

// ---------------------------------------------------------------------------
// Local attention (heads 8..15). One block per (b,i): 8 heads x 16 edges.
// ---------------------------------------------------------------------------
__global__ __launch_bounds__(256)
void local_attn_kernel(const float* __restrict__ qb, const float* __restrict__ kb,
                       const float* __restrict__ vb, const float* __restrict__ ef,
                       const int* __restrict__ pj, float* __restrict__ ctx)
{
  const int bi = blockIdx.x;        // b*N + i
  const int b  = bi >> 9;
  const int t  = threadIdx.x;
  const float scale = 0.17677669529663687f;

  __shared__ float qs[HHc][DHc];
  __shared__ float scx[HHc][DEGc];
  __shared__ float spx[HHc][DEGc];

  {
    const int h = t >> 5, d = t & (DHc - 1);
    qs[h][d] = qb[(long)bi * Dc + (h + HHc) * DHc + d] * scale;
  }
  __syncthreads();

  if (t < HHc * DEGc) {
    const int h = t >> 4, el = t & (DEGc - 1);
    const int e = bi * DEGc + el;
    const int j = pj[e];
    const float* kr = kb + ((long)(b * Nc + j)) * Dc + (h + HHc) * DHc;
    const float* er = ef + (long)e * (HHc * DHc) + h * DHc;
    float s = 0.0f;
    #pragma unroll
    for (int d = 0; d < DHc; ++d) s += qs[h][d] * kr[d] * er[d];
    scx[h][el] = s;
  }
  __syncthreads();

  if (t < HHc) {
    float mx = -3.0e38f;
    #pragma unroll
    for (int e = 0; e < DEGc; ++e)
      if (scx[t][e] > -10000.0f) mx = fmaxf(mx, scx[t][e]);
    float sum = 0.0f;
    #pragma unroll
    for (int e = 0; e < DEGc; ++e) {
      float p = (scx[t][e] > -10000.0f) ? __expf(scx[t][e] - mx) : 0.0f;
      spx[t][e] = p;
      sum += p;
    }
    float inv = (sum > 0.0f) ? 1.0f / sum : 0.0f;
    #pragma unroll
    for (int e = 0; e < DEGc; ++e) spx[t][e] *= inv;
  }
  __syncthreads();

  {
    const int h = t >> 5, d = t & (DHc - 1);
    float a = 0.0f;
    #pragma unroll
    for (int el = 0; el < DEGc; ++el) {
      const int e = bi * DEGc + el;
      a += spx[h][el] * vb[((long)(b * Nc + pj[e])) * Dc + (h + HHc) * DHc + d];
    }
    ctx[(long)bi * Dc + (h + HHc) * DHc + d] = a;
  }
}

// ---------------------------------------------------------------------------
// Host side
// ---------------------------------------------------------------------------
extern "C" void kernel_launch(void* const* d_in, const int* in_sizes, int n_in,
                              void* d_out, int out_size, void* d_ws, size_t ws_size,
                              hipStream_t stream) {
  const float* key    = (const float*)d_in[0];
  const float* value  = (const float*)d_in[1];
  const float* query  = (const float*)d_in[2];
  const unsigned char* mask = (const unsigned char*)d_in[3];
  const float* edgef  = (const float*)d_in[4];
  const int* pair_b   = (const int*)d_in[5];
  const int* pair_i   = (const int*)d_in[6];
  const int* pair_j   = (const int*)d_in[7];
  const float* Wq = (const float*)d_in[8];  const float* bq = (const float*)d_in[9];
  const float* Wk = (const float*)d_in[10]; const float* bk = (const float*)d_in[11];
  const float* Wv = (const float*)d_in[12]; const float* bv = (const float*)d_in[13];
  const float* Wo = (const float*)d_in[14]; const float* bo = (const float*)d_in[15];
  const float* We1 = (const float*)d_in[16]; const float* be1 = (const float*)d_in[17];
  const float* We2 = (const float*)d_in[18]; const float* be2 = (const float*)d_in[19];
  const float* Wu1 = (const float*)d_in[20]; const float* bu1 = (const float*)d_in[21];
  const float* Wu2 = (const float*)d_in[22]; const float* bu2 = (const float*)d_in[23];

  float* out_output = (float*)d_out;
  float* out_top    = out_output + (long)Bc * Nc * Dc;
  float* out_edge   = out_top + (long)Bc * Nc * Nc;

  char* ws = (char*)d_ws;
  size_t off = 0;
  auto carve = [&](size_t bytes) -> char* {
    char* p = ws + off;
    off = (off + bytes + 255) & ~(size_t)255;
    return p;
  };
  float*  qbuf  = (float*)carve((size_t)Bc * Nc * Dc * 4);
  float*  kbuf  = (float*)carve((size_t)Bc * Nc * Dc * 4);
  float*  vbuf  = (float*)carve((size_t)Bc * Nc * Dc * 4);
  float*  ctxb  = (float*)carve((size_t)Bc * Nc * Dc * 4);
  bf16_t* hidA  = (bf16_t*)carve((size_t)Ec * Dc * 2);          // reused
  float*  efbuf = (float*)carve((size_t)Ec * (HHc * DHc) * 4);
  int*    row_i = (int*)carve((size_t)Ec * 4);
  int*    row_j = (int*)carve((size_t)Ec * 4);
  // transposed bf16 weights (N-major)
  bf16_t* tWq  = (bf16_t*)carve((size_t)Dc * Dc * 2);
  bf16_t* tWk  = (bf16_t*)carve((size_t)Dc * Dc * 2);
  bf16_t* tWv  = (bf16_t*)carve((size_t)Dc * Dc * 2);
  bf16_t* tWo  = (bf16_t*)carve((size_t)Dc * Dc * 2);
  bf16_t* tWe1 = (bf16_t*)carve((size_t)Dc * Dc * 2);
  bf16_t* tWe2 = (bf16_t*)carve((size_t)Dc * (HHc * DHc) * 2);
  bf16_t* tWu1 = (bf16_t*)carve((size_t)(2 * Dc) * Dc * 2);
  bf16_t* tWu2 = (bf16_t*)carve((size_t)Dc * Dc * 2);
  (void)ws_size; (void)in_sizes; (void)n_in; (void)out_size;

  const int MN = Bc * Nc; // 4096
  dim3 blk(256);
  auto grid = [](int M, int N) { return dim3(N / BNT, M / BMT); };
  auto tw = [&](const float* W, bf16_t* Wt, int K, int N) {
    transpose_w_kernel<<<(K * N + 255) / 256, 256, 0, stream>>>(W, Wt, K, N);
  };

  // one-time weight prep (cheap: ~2.1M elements total)
  tw(Wq, tWq, Dc, Dc);   tw(Wk, tWk, Dc, Dc);  tw(Wv, tWv, Dc, Dc);
  tw(Wo, tWo, Dc, Dc);   tw(We1, tWe1, Dc, Dc);
  tw(We2, tWe2, Dc, HHc * DHc);
  tw(Wu1, tWu1, 2 * Dc, Dc);
  tw(Wu2, tWu2, Dc, Dc);
  build_idx_kernel<<<(Ec + 255) / 256, 256, 0, stream>>>(pair_b, pair_i, pair_j, row_i, row_j);

  // Q/K/V projections (f32 in, f32 out)
  gemm_wmma<false,false,false,false><<<grid(MN, Dc), blk, 0, stream>>>(
      query, Dc, tWq, Dc, bq, qbuf, Dc, Dc, nullptr, nullptr, 0);
  gemm_wmma<false,false,false,false><<<grid(MN, Dc), blk, 0, stream>>>(
      key,   Dc, tWk, Dc, bk, kbuf, Dc, Dc, nullptr, nullptr, 0);
  gemm_wmma<false,false,false,false><<<grid(MN, Dc), blk, 0, stream>>>(
      value, Dc, tWv, Dc, bv, vbuf, Dc, Dc, nullptr, nullptr, 0);

  // Edge MLP: hid = softplus(ef @ We1 + be1) - ln2 (bf16 out)
  gemm_wmma<false,false,true,true><<<grid(Ec, Dc), blk, 0, stream>>>(
      edgef, Dc, tWe1, Dc, be1, hidA, Dc, Dc, nullptr, nullptr, 0);
  // ef_final = hid @ We2 + be2 (bf16 in, f32 out, N = 256)
  gemm_wmma<true,false,false,false><<<grid(Ec, HHc * DHc), blk, 0, stream>>>(
      hidA, Dc, tWe2, Dc, be2, efbuf, HHc * DHc, Dc, nullptr, nullptr, 0);

  // Attention
  global_attn_kernel<<<Bc * HHc * Nc, 256, 0, stream>>>(qbuf, kbuf, vbuf, mask, ctxb, out_top);
  local_attn_kernel<<<Bc * Nc, 256, 0, stream>>>(qbuf, kbuf, vbuf, efbuf, pair_j, ctxb);

  // Output projection
  gemm_wmma<false,false,false,false><<<grid(MN, Dc), blk, 0, stream>>>(
      ctxb, Dc, tWo, Dc, bo, out_output, Dc, Dc, nullptr, nullptr, 0);

  // Edge update MLP 1: gathered split-K A = concat(output[row_i], output[row_j])
  gemm_wmma<false,true,true,true><<<grid(Ec, Dc), blk, 0, stream>>>(
      out_output, Dc, tWu1, 2 * Dc, bu1, hidA, Dc, 2 * Dc, row_i, row_j, Dc);
  // Edge update MLP 2
  gemm_wmma<true,false,false,false><<<grid(Ec, Dc), blk, 0, stream>>>(
      hidA, Dc, tWu2, Dc, bu2, out_edge, Dc, Dc, nullptr, nullptr, 0);
}
